// LSTMCell_scratch_76055280877767
// MI455X (gfx1250) — matrix operand order repaired and checked by reading, not compiled, not run on hardware
//
#include <hip/hip_runtime.h>
#include <hip/hip_bf16.h>
#include <math.h>

// ---------------------------------------------------------------------------
// LSTM cell, fused 4-gate GEMM on CDNA5 WMMA (bf16 in, f32 accumulate)
//   combined = [x | h_prev]  : (16384, 2048) f32
//   gate_z   = combined @ Wz^T + bz, z in {f,i,g,o}
//   c_next = sigmoid(f)*c_prev + sigmoid(i)*tanh(g)
//   h_next = sigmoid(o)*tanh(c_next)
// ---------------------------------------------------------------------------

typedef __attribute__((ext_vector_type(16))) __bf16 v16bf;
typedef __attribute__((ext_vector_type(8)))  float  v8f;

#define B_ROWS 16384
#define IN_DIM 1024
#define H_DIM  1024
#define K_DIM  2048   // IN_DIM + H_DIM

#define BM 128        // block tile rows (batch)
#define BN 64         // block tile cols (hidden) per gate
#define BK 64         // K per LDS stage (2 WMMA k-steps)
#define LDK 72        // padded LDS row stride in bf16 elems (144B, 16B-aligned)

__device__ __forceinline__ float sigmoid_(float v) {
    return 1.0f / (1.0f + __expf(-v));
}

__global__ __launch_bounds__(256, 1)
void lstm_cell_wmma_bf16(const float* __restrict__ x,
                         const float* __restrict__ h_prev,
                         const float* __restrict__ c_prev,
                         const float* __restrict__ Wf, const float* __restrict__ bf_,
                         const float* __restrict__ Wi, const float* __restrict__ bi_,
                         const float* __restrict__ Wg, const float* __restrict__ bg_,
                         const float* __restrict__ Wo, const float* __restrict__ bo_,
                         float* __restrict__ out_h,
                         float* __restrict__ out_c)
{
    __shared__ __align__(16) __bf16 As[BM][LDK];        // combined tile (row-major, k contiguous)
    __shared__ __align__(16) __bf16 Bs[4][BN][LDK];     // per-gate weight tile W[n][k]

    const int tid  = threadIdx.x;
    const int lane = tid & 31;
    const int wave = tid >> 5;       // 0..7
    const int wm   = wave >> 1;      // 0..3 : 32-row slab
    const int wn   = wave & 1;       // 0..1 : 32-col slab

    const int row0 = blockIdx.y * BM;
    const int col0 = blockIdx.x * BN;

    const float* Wp[4] = {Wf, Wi, Wg, Wo};

    v8f acc[4][2][2];                // [gate][mi][ni], 16x16 f32 tiles
#pragma unroll
    for (int gi = 0; gi < 4; ++gi)
#pragma unroll
        for (int mi = 0; mi < 2; ++mi)
#pragma unroll
            for (int ni = 0; ni < 2; ++ni)
                acc[gi][mi][ni] = (v8f){0.f,0.f,0.f,0.f,0.f,0.f,0.f,0.f};

    const int grp = lane >> 4;       // 0/1 : lane group
    const int l16 = lane & 15;       // M row (A) / N col (B) within 16

    for (int kt = 0; kt < K_DIM / BK; ++kt) {
        const int k0 = kt * BK;
        // concat(x, h_prev): stage is entirely inside one source (1024 % 64 == 0)
        const float* srcA = (k0 < IN_DIM)
            ? (x      + (size_t)row0 * IN_DIM + k0)
            : (h_prev + (size_t)row0 * H_DIM  + (k0 - IN_DIM));

        __syncthreads();

        // ---- stage A tile: 128x64 f32 -> bf16, 8 x float4 per thread ----
#pragma unroll
        for (int i = 0; i < 8; ++i) {
            const int linear = i * 256 + tid;        // 0..2047 float4 slots
            const int r  = linear >> 4;              // 16 float4 per row
            const int kq = (linear & 15) << 2;
            const float4 v = *(const float4*)(srcA + (size_t)r * 1024 + kq);
            As[r][kq + 0] = (__bf16)v.x;
            As[r][kq + 1] = (__bf16)v.y;
            As[r][kq + 2] = (__bf16)v.z;
            As[r][kq + 3] = (__bf16)v.w;
        }

        // ---- stage B tiles: 4 x (64x64) f32 -> bf16 ----
#pragma unroll
        for (int gi = 0; gi < 4; ++gi) {
            const float* srcB = Wp[gi] + (size_t)col0 * K_DIM + k0;
#pragma unroll
            for (int i = 0; i < 4; ++i) {
                const int linear = i * 256 + tid;    // 0..1023 float4 slots
                const int n  = linear >> 4;
                const int kq = (linear & 15) << 2;
                const float4 v = *(const float4*)(srcB + (size_t)n * K_DIM + kq);
                Bs[gi][n][kq + 0] = (__bf16)v.x;
                Bs[gi][n][kq + 1] = (__bf16)v.y;
                Bs[gi][n][kq + 2] = (__bf16)v.z;
                Bs[gi][n][kq + 3] = (__bf16)v.w;
            }
        }

        __syncthreads();

        // ---- compute: 2 k-steps of 32, 4 gates, 2x2 16x16 tiles per wave ----
#pragma unroll
        for (int ks = 0; ks < 2; ++ks) {
            // A fragment (16x32 bf16): lane(group g, m): halves h -> K = g*16+h.
            // 16 contiguous halves = two contiguous 16B chunks.
            const int kbA = ks * 32 + grp * 16;
            v16bf afrag[2];
#pragma unroll
            for (int mi = 0; mi < 2; ++mi) {
                const __bf16* ap = &As[wm * 32 + mi * 16 + l16][0];
#pragma unroll
                for (int h = 0; h < 8; ++h) {
                    afrag[mi][h]     = ap[kbA + h];
                    afrag[mi][h + 8] = ap[kbA + 8 + h];
                }
            }

            // B fragment (32x16 bf16): lane(group g, n): halves h -> K = g*16+h.
            // 16 contiguous halves = 32B = two ds_load_b128.
            const int kbB = ks * 32 + grp * 16;
#pragma unroll
            for (int gi = 0; gi < 4; ++gi) {
#pragma unroll
                for (int ni = 0; ni < 2; ++ni) {
                    const __bf16* bp = &Bs[gi][wn * 32 + ni * 16 + l16][0];
                    v16bf bfrag;
#pragma unroll
                    for (int h = 0; h < 16; ++h) bfrag[h] = bp[kbB + h];
#pragma unroll
                    for (int mi = 0; mi < 2; ++mi) {
                        acc[gi][mi][ni] = __builtin_amdgcn_wmma_f32_16x16x32_bf16(
                            /*neg_a=*/false, afrag[mi],
                            /*neg_b=*/false, bfrag,
                            /*c_mod=*/(short)0, acc[gi][mi][ni],
                            /*reuse_a=*/false, /*reuse_b=*/false);
                    }
                }
            }
        }
    }

    // ---- epilogue: bias + activations + cell update, in-register ----
    // C/D layout: lanes 0-15 -> M = v, N = lane; lanes 16-31 -> M = 8+v, N = lane-16
#pragma unroll
    for (int ni = 0; ni < 2; ++ni) {
        const int colg  = col0 + wn * 32 + ni * 16 + l16;
        const float bfv = bf_[colg];
        const float biv = bi_[colg];
        const float bgv = bg_[colg];
        const float bov = bo_[colg];
#pragma unroll
        for (int mi = 0; mi < 2; ++mi) {
#pragma unroll
            for (int v = 0; v < 8; ++v) {
                const int rowg = row0 + wm * 32 + mi * 16 + grp * 8 + v;
                const size_t idx = (size_t)rowg * H_DIM + colg;
                const float fv = sigmoid_(acc[0][mi][ni][v] + bfv);
                const float iv = sigmoid_(acc[1][mi][ni][v] + biv);
                const float gv = tanhf   (acc[2][mi][ni][v] + bgv);
                const float ov = sigmoid_(acc[3][mi][ni][v] + bov);
                const float cp = c_prev[idx];
                const float cn = fv * cp + iv * gv;
                out_h[idx] = ov * tanhf(cn);
                out_c[idx] = cn;
            }
        }
    }
}

extern "C" void kernel_launch(void* const* d_in, const int* in_sizes, int n_in,
                              void* d_out, int out_size, void* d_ws, size_t ws_size,
                              hipStream_t stream) {
    (void)in_sizes; (void)n_in; (void)out_size; (void)d_ws; (void)ws_size;
    const float* x      = (const float*)d_in[0];
    const float* h_prev = (const float*)d_in[1];
    const float* c_prev = (const float*)d_in[2];
    const float* Wf     = (const float*)d_in[3];
    const float* bf_    = (const float*)d_in[4];
    const float* Wi     = (const float*)d_in[5];
    const float* bi_    = (const float*)d_in[6];
    const float* Wg     = (const float*)d_in[7];
    const float* bg_    = (const float*)d_in[8];
    const float* Wo     = (const float*)d_in[9];
    const float* bo_    = (const float*)d_in[10];

    float* out_h = (float*)d_out;                          // h_next first
    float* out_c = out_h + (size_t)B_ROWS * H_DIM;         // then c_next

    dim3 grid(H_DIM / BN, B_ROWS / BM, 1);                 // (16, 128)
    lstm_cell_wmma_bf16<<<grid, dim3(256, 1, 1), 0, stream>>>(
        x, h_prev, c_prev, Wf, bf_, Wi, bi_, Wg, bg_, Wo, bo_, out_h, out_c);
}
